// SimpleRNN_65979287601595
// MI455X (gfx1250) — matrix operand order repaired and checked by reading, not compiled, not run on hardware
//
#include <hip/hip_runtime.h>
#include <cstdint>
#include <cstddef>

#define HID   512
#define BATCH 64
#define STEPS 2048
#define MTOT  (STEPS * BATCH)   // 131072 rows of the x_proj GEMM

typedef __attribute__((ext_vector_type(16))) _Float16 v16h;
typedef __attribute__((ext_vector_type(8)))  _Float16 v8h;
typedef __attribute__((ext_vector_type(8)))  float    v8f;

// Workspace layout (bytes):
//   [0,      512K) : Wx f16 [n][k] = W[n][512+k]   (B-matrix for x_proj GEMM, col n contiguous)
//   [512K,   1M  ) : Wh f16 [n][k] = W[n][k]       (B-matrix for recurrence)
//   [1M,     +64K) : h ping  f16 [64][512]
//   [+64K,  +128K) : h pong  f16 [64][512]
//   [+128K]        : barrier counter (unsigned)
//   [2M, 2M+256M)  : x_proj f32 [t][b][h]

__device__ __forceinline__ void loadA8_f32(const float* p, v16h& a, int o) {
    float4 x = ((const float4*)p)[0];
    float4 y = ((const float4*)p)[1];
    a[o+0] = (_Float16)x.x; a[o+1] = (_Float16)x.y; a[o+2] = (_Float16)x.z; a[o+3] = (_Float16)x.w;
    a[o+4] = (_Float16)y.x; a[o+5] = (_Float16)y.y; a[o+6] = (_Float16)y.z; a[o+7] = (_Float16)y.w;
}

// ---------------- prep: split/convert W, zero h0, reset barrier ----------------
__global__ __launch_bounds__(256) void prep_kernel(const float* __restrict__ W,
                                                   _Float16* __restrict__ Wx,
                                                   _Float16* __restrict__ Wh,
                                                   _Float16* __restrict__ h0,
                                                   unsigned* __restrict__ counter) {
    int idx = blockIdx.x * blockDim.x + threadIdx.x;   // 0 .. 512*512-1
    int n = idx >> 9, k = idx & 511;
    Wh[idx] = (_Float16)W[n * 1024 + k];
    Wx[idx] = (_Float16)W[n * 1024 + 512 + k];
    if (idx < BATCH * HID) h0[idx] = (_Float16)0.0f;
    if (idx == 0) *counter = 0u;
}

// ---------------- phase 1: x_proj = inputs @ Wx^T + b  (bandwidth-bound) ----------------
// Block = 256 thr = 8 waves; block owns one 16-row M strip; wave w owns N columns [w*64, w*64+64).
__global__ __launch_bounds__(256) void xproj_gemm(const float*    __restrict__ A,
                                                  const _Float16* __restrict__ Wx,
                                                  const float*    __restrict__ bias,
                                                  float*          __restrict__ C) {
    const int wave = threadIdx.x >> 5, lane = threadIdx.x & 31;
    const int lr = lane & 15, hi = lane >> 4;
    const int mrow = blockIdx.x * 16 + lr;            // A row for this lane's fragment
    const float* arow = A + (size_t)mrow * HID;

    v8f acc[4] = {};
#pragma unroll
    for (int kt = 0; kt < 16; ++kt) {
        v16h a;
        loadA8_f32(arow + kt * 32 + hi * 16,     a, 0);   // K slots 0..7
        loadA8_f32(arow + kt * 32 + hi * 16 + 8, a, 8);   // K slots 8..15
#pragma unroll
        for (int nt = 0; nt < 4; ++nt) {
            const int ncol = wave * 64 + nt * 16 + lr;    // B column for this lane
            v16h b = *(const v16h*)(Wx + (size_t)ncol * HID + kt * 32 + hi * 16);
            acc[nt] = __builtin_amdgcn_wmma_f32_16x16x32_f16(
                false, a, false, b, (short)0, acc[nt], false, false);
        }
    }
#pragma unroll
    for (int nt = 0; nt < 4; ++nt) {
        const int col = wave * 64 + nt * 16 + lr;
        const float bv = bias[col];
#pragma unroll
        for (int r = 0; r < 8; ++r) {
            const int row = blockIdx.x * 16 + r + 8 * hi;
            C[(size_t)row * HID + col] = acc[nt][r] + bv;
        }
    }
}

// ---------------- phase 2: persistent recurrent scan ----------------
// 16 blocks (one per WGP) x 256 thr (8 waves). Block owns a 32-column slice of h_new.
// Wave (mt = w&3, nt = w>>2) computes output tile (M rows mt*16.., N cols blk*32+nt*16..).
// Its 512x16 slice of W_h^T lives in 128 VGPRs for the whole kernel.
// K=512 is accumulated as TWO independent 8-deep WMMA chains to halve exposed latency.
__global__ __launch_bounds__(256) void rnn_scan(const float*    __restrict__ xp,
                                                const _Float16* __restrict__ Wh,
                                                _Float16* __restrict__ h0,
                                                _Float16* __restrict__ h1,
                                                unsigned* __restrict__ counter,
                                                float* __restrict__ out) {
    const int wave = threadIdx.x >> 5, lane = threadIdx.x & 31;
    const int lr = lane & 15, hi = lane >> 4;
    const int mt = wave & 3;                      // M tile 0..3 (batch rows)
    const int nt = wave >> 2;                     // local N tile 0..1
    const int ncol  = blockIdx.x * 32 + nt * 16 + lr;   // output column / B column
    const int mbase = mt * 16;
    const int arow  = mbase + lr;                 // A row for fragments

    // Preload this wave's W_h slice into registers: 16 k-tiles x 8 VGPRs = 128 VGPRs.
    v16h Bf[16];
#pragma unroll
    for (int kt = 0; kt < 16; ++kt)
        Bf[kt] = *(const v16h*)(Wh + (size_t)ncol * HID + kt * 32 + hi * 16);

    unsigned tgt = 0;
    for (int t = 0; t < STEPS; ++t) {
        const _Float16* hin  = (t & 1) ? h1 : h0;
        _Float16*       hout = (t & 1) ? h0 : h1;

        // x_proj for this tile (independent of h -> overlaps the WMMA chains),
        // plus prefetch of next step's slice to keep HBM latency off the critical path.
        float xv[8];
#pragma unroll
        for (int r = 0; r < 8; ++r) {
            const int row = mbase + r + 8 * hi;
            xv[r] = xp[((size_t)t * BATCH + row) * HID + ncol];
            if (t + 1 < STEPS)
                __builtin_prefetch(&xp[((size_t)(t + 1) * BATCH + row) * HID + ncol], 0, 0);
        }

        // y = h @ W_h^T : two independent 8-deep chains over K=512.
        v8f acc0 = {}, acc1 = {};
#pragma unroll
        for (int kt = 0; kt < 8; ++kt) {
            {
                const _Float16* pa = hin + (size_t)arow * HID + kt * 32 + hi * 16;
                v8h lo  = *(const v8h*)pa;
                v8h hi8 = *(const v8h*)(pa + 8);
                v16h a;
#pragma unroll
                for (int j = 0; j < 8; ++j) { a[j] = lo[j]; a[8 + j] = hi8[j]; }
                acc0 = __builtin_amdgcn_wmma_f32_16x16x32_f16(
                    false, a, false, Bf[kt], (short)0, acc0, false, false);
            }
            {
                const _Float16* pa = hin + (size_t)arow * HID + (kt + 8) * 32 + hi * 16;
                v8h lo  = *(const v8h*)pa;
                v8h hi8 = *(const v8h*)(pa + 8);
                v16h a;
#pragma unroll
                for (int j = 0; j < 8; ++j) { a[j] = lo[j]; a[8 + j] = hi8[j]; }
                acc1 = __builtin_amdgcn_wmma_f32_16x16x32_f16(
                    false, a, false, Bf[kt + 8], (short)0, acc1, false, false);
            }
        }

        // h_new = tanh(y + xp); publish f16 slice; final step also writes f32 output.
#pragma unroll
        for (int r = 0; r < 8; ++r) {
            const int row = mbase + r + 8 * hi;
            const float v = tanhf(acc0[r] + acc1[r] + xv[r]);
            hout[row * HID + ncol] = (_Float16)v;
            if (t == STEPS - 1) out[(size_t)row * HID + ncol] = v;
        }

        // Device-wide barrier across the 16 persistent blocks.
        __threadfence();
        __syncthreads();
        tgt += gridDim.x;
        if (threadIdx.x == 0) {
            __hip_atomic_fetch_add(counter, 1u, __ATOMIC_RELEASE, __HIP_MEMORY_SCOPE_AGENT);
            while (__hip_atomic_load(counter, __ATOMIC_ACQUIRE, __HIP_MEMORY_SCOPE_AGENT) < tgt)
                __builtin_amdgcn_s_sleep(1);
        }
        __syncthreads();
        __threadfence();
    }
}

extern "C" void kernel_launch(void* const* d_in, const int* in_sizes, int n_in,
                              void* d_out, int out_size, void* d_ws, size_t ws_size,
                              hipStream_t stream) {
    const float* inputs = (const float*)d_in[0];   // (2048, 64, 512) f32
    const float* W      = (const float*)d_in[1];   // (512, 1024)     f32
    const float* bias   = (const float*)d_in[2];   // (512,)          f32
    float* out = (float*)d_out;                    // (64, 512)       f32

    char* ws = (char*)d_ws;
    _Float16* Wx      = (_Float16*)(ws);
    _Float16* Wh      = (_Float16*)(ws + (512 << 10));
    _Float16* h0      = (_Float16*)(ws + (1 << 20));
    _Float16* h1      = (_Float16*)(ws + (1 << 20) + (64 << 10));
    unsigned* counter = (unsigned*)(ws + (1 << 20) + (128 << 10));
    float*    xp      = (float*)(ws + (2 << 20));

    prep_kernel<<<1024, 256, 0, stream>>>(W, Wx, Wh, h0, counter);
    xproj_gemm<<<MTOT / 16, 256, 0, stream>>>(inputs, Wx, bias, xp);
    rnn_scan<<<16, 256, 0, stream>>>(xp, Wh, h0, h1, counter, out);
}
